// StructTokenMemory_77378130804782
// MI455X (gfx1250) — matrix operand order repaired and checked by reading, not compiled, not run on hardware
//
#include <hip/hip_runtime.h>
#include <stdint.h>

#define IN_DIM      128
#define TOKEN_DIM   128
#define NUM_TOKENS  256
#define TOPK        8
#define NSEG        512
#define TILE_ROWS   128
#define RSQRT_D     0.08838834764831845f

typedef unsigned short u16;
typedef __attribute__((ext_vector_type(16))) __bf16 v16bf;
typedef __attribute__((ext_vector_type(8)))  float  v8f;

union Frag {
    v16bf bf;
    uint4 q[2];
};

__device__ __forceinline__ u16 f2bf(float x) {
    unsigned int u = __float_as_uint(x);
    u += 0x7FFFu + ((u >> 16) & 1u);   // round-to-nearest-even
    return (u16)(u >> 16);
}

// ---------------------------------------------------------------------------
// Pack W_q and tokens_K into WMMA B-matrix fragment order (bf16).
// B-frag layout (16x16x32 bf16): lane L -> column n = L%16, elem j -> k = (L/16)*8 + (j&7) + (j>>3)*16
// Stored as [kt][nt][lane][j] so each lane loads 32 contiguous bytes.
// ---------------------------------------------------------------------------
__global__ void pack_kernel(const float* __restrict__ W_q,
                            const float* __restrict__ tokens_K,
                            u16* __restrict__ WqB, u16* __restrict__ KtB) {
    int i = blockIdx.x * blockDim.x + threadIdx.x;   // 32768 threads
    if (i < 16384) { // W_q: 4 ktiles x 8 ntiles x 32 lanes x 16
        int j    = i & 15;
        int lane = (i >> 4) & 31;
        int nt   = (i >> 9) & 7;
        int kt   = i >> 12;
        int n = nt * 16 + (lane & 15);
        int k = kt * 32 + (lane >> 4) * 8 + (j & 7) + (j >> 3) * 16;
        WqB[i] = f2bf(W_q[n * IN_DIM + k]);          // B[k][n] = W_q[n][k]
    }
    if (i < 32768) { // tokens_K: 4 ktiles x 16 ntiles x 32 lanes x 16
        int j    = i & 15;
        int lane = (i >> 4) & 31;
        int nt   = (i >> 9) & 15;
        int kt   = i >> 13;
        int n = nt * 16 + (lane & 15);
        int k = kt * 32 + (lane >> 4) * 8 + (j & 7) + (j >> 3) * 16;
        KtB[i] = f2bf(tokens_K[n * TOKEN_DIM + k]);
    }
}

// ---------------------------------------------------------------------------
// Main kernel: per 128-row tile: Q = Xbf16 @ WqT + bq (WMMA), S = Qbf16 @ KT (WMMA),
// top-8 + softmax per row, scatter routing weights into Wacc[seg][token].
// ---------------------------------------------------------------------------
__global__ __launch_bounds__(256)
void main_kernel(const float* __restrict__ v_emb,
                 const int*   __restrict__ batch_idx,
                 const float* __restrict__ b_q,
                 const u16*   __restrict__ WqB,
                 const u16*   __restrict__ KtB,
                 float* __restrict__ Wacc,
                 float* __restrict__ counts,
                 int Nrows) {
    extern __shared__ char smem[];
    u16*   sQ  = (u16*)smem;            // 128x128 bf16 Q           (32 KB)
    u16*   sX  = sQ  + 16384;           // 128x128 bf16 X           (32 KB)
    u16*   sWq = sX  + 16384;           // Wq  B-frag order         (32 KB)
    u16*   sK  = sWq + 16384;           // K   B-frag order         (64 KB)
    float* sS  = (float*)(sK + 32768);  // 128x256 f32 scores      (128 KB)
    float* sBq = sS + 128 * NUM_TOKENS; // 128 f32 bias            (512 B)

    const int tid  = threadIdx.x;
    const int wave = tid >> 5;
    const int lane = tid & 31;
    const int mt   = wave;              // each wave owns one 16-row M-tile
    const int r0   = blockIdx.x * TILE_ROWS;

    // ---- cooperative stage-in --------------------------------------------
    {
        const uint4* g = (const uint4*)WqB;
        uint4* l = (uint4*)sWq;
        for (int i = tid; i < 2048; i += 256) l[i] = g[i];
    }
    {
        const uint4* g = (const uint4*)KtB;
        uint4* l = (uint4*)sK;
        for (int i = tid; i < 4096; i += 256) l[i] = g[i];
    }
    if (tid < TOKEN_DIM) sBq[tid] = b_q[tid];
    {
        const float4* gX = (const float4*)(v_emb + (size_t)r0 * IN_DIM);
        for (int i = tid; i < 4096; i += 256) {
            int e    = i * 4;
            int grow = r0 + (e >> 7);
            float4 v = make_float4(0.f, 0.f, 0.f, 0.f);
            if (grow < Nrows) v = gX[i];
            uint2 p;
            p.x = (unsigned)f2bf(v.x) | ((unsigned)f2bf(v.y) << 16);
            p.y = (unsigned)f2bf(v.z) | ((unsigned)f2bf(v.w) << 16);
            ((uint2*)sX)[i] = p;
        }
    }
    __syncthreads();

    const int hf = lane >> 4;     // which K-half this lane holds
    const int ln = lane & 15;

    // ---- GEMM1: Q[128x128] = X @ Wq^T + b_q ------------------------------
    Frag afr[4];
#pragma unroll
    for (int kt = 0; kt < 4; ++kt) {
        const u16* base = sX + (mt * 16 + ln) * IN_DIM + kt * 32 + hf * 8;
        afr[kt].q[0] = *(const uint4*)(base);
        afr[kt].q[1] = *(const uint4*)(base + 16);
    }
    for (int nt = 0; nt < 8; ++nt) {
        float bias = sBq[nt * 16 + ln];
        v8f acc;
#pragma unroll
        for (int r = 0; r < 8; ++r) acc[r] = bias;   // C[m][n] = b_q[n]
#pragma unroll
        for (int kt = 0; kt < 4; ++kt) {
            Frag bfr;
            const uint4* bs = (const uint4*)(sWq + (((kt * 8 + nt) * 32 + lane) << 4));
            bfr.q[0] = bs[0];
            bfr.q[1] = bs[1];
            acc = __builtin_amdgcn_wmma_f32_16x16x32_bf16(
                false, afr[kt].bf, false, bfr.bf, (short)0, acc, false, false);
        }
#pragma unroll
        for (int r = 0; r < 8; ++r) {
            int m = mt * 16 + r + 8 * hf;
            sQ[m * TOKEN_DIM + nt * 16 + ln] = f2bf(acc[r]);
        }
    }

    // ---- GEMM2: S[128x256] = Qbf16 @ K^T, scaled -------------------------
    Frag qfr[4];
#pragma unroll
    for (int kt = 0; kt < 4; ++kt) {
        const u16* base = sQ + (mt * 16 + ln) * TOKEN_DIM + kt * 32 + hf * 8;
        qfr[kt].q[0] = *(const uint4*)(base);
        qfr[kt].q[1] = *(const uint4*)(base + 16);
    }
    for (int nt = 0; nt < 16; ++nt) {
        v8f acc;
#pragma unroll
        for (int r = 0; r < 8; ++r) acc[r] = 0.f;
#pragma unroll
        for (int kt = 0; kt < 4; ++kt) {
            Frag bfr;
            const uint4* bs = (const uint4*)(sK + (((kt * 16 + nt) * 32 + lane) << 4));
            bfr.q[0] = bs[0];
            bfr.q[1] = bs[1];
            acc = __builtin_amdgcn_wmma_f32_16x16x32_bf16(
                false, qfr[kt].bf, false, bfr.bf, (short)0, acc, false, false);
        }
#pragma unroll
        for (int r = 0; r < 8; ++r) {
            int m = mt * 16 + r + 8 * hf;
            sS[m * NUM_TOKENS + nt * 16 + ln] = acc[r] * RSQRT_D;
        }
    }

    // ---- top-8 + softmax + sparse scatter (wave-cooperative, own rows) ----
    for (int rr = 0; rr < 16; ++rr) {
        int row  = mt * 16 + rr;
        int grow = r0 + row;
        if (grow >= Nrows) continue;

        const float* srow = sS + row * NUM_TOKENS;
        float vals[8];
#pragma unroll
        for (int j = 0; j < 8; ++j) vals[j] = srow[lane * 8 + j];

        int seg = batch_idx[grow];

        float maxv0 = 0.f, esum = 0.f;
        int   myTok[TOPK];
        float myE[TOPK];
        int   myCnt = 0;

        for (int it = 0; it < TOPK; ++it) {
            float lm = -__builtin_inff();
            int   lj = 0;
#pragma unroll
            for (int j = 0; j < 8; ++j)
                if (vals[j] > lm) { lm = vals[j]; lj = j; }
            float wm = lm;
#pragma unroll
            for (int off = 16; off > 0; off >>= 1)
                wm = fmaxf(wm, __shfl_xor(wm, off, 32));
            if (it == 0) maxv0 = wm;
            float e = __expf(wm - maxv0);
            esum += e;
            unsigned int mask = __builtin_amdgcn_ballot_w32(lm == wm);
            int src = __ffs(mask) - 1;
            if (lane == src) {
                vals[lj]      = -__builtin_inff();
                myTok[myCnt]  = lane * 8 + lj;
                myE[myCnt]    = e;
                myCnt++;
            }
        }
        float inv = 1.0f / esum;
        for (int c = 0; c < myCnt; ++c)
            atomicAdd(&Wacc[(size_t)seg * NUM_TOKENS + myTok[c]], myE[c] * inv);
        if (lane == 0)
            atomicAdd(&counts[seg], 1.0f);
    }
}

// ---------------------------------------------------------------------------
// Finish: struct_emb[b,d] = (Wacc[b,:] @ tokens_V[:,d]) / max(count[b],1)
// ---------------------------------------------------------------------------
__global__ __launch_bounds__(TOKEN_DIM)
void finish_kernel(const float* __restrict__ Wacc,
                   const float* __restrict__ counts,
                   const float* __restrict__ tokens_V,
                   float* __restrict__ out) {
    int b = blockIdx.x;
    int d = threadIdx.x;
    const float* wrow = Wacc + (size_t)b * NUM_TOKENS;
    float s = 0.f;
#pragma unroll 8
    for (int t = 0; t < NUM_TOKENS; ++t)
        s += wrow[t] * tokens_V[t * TOKEN_DIM + d];
    float c = counts[b];
    out[b * TOKEN_DIM + d] = s / fmaxf(c, 1.0f);
}

// ---------------------------------------------------------------------------
extern "C" void kernel_launch(void* const* d_in, const int* in_sizes, int n_in,
                              void* d_out, int out_size, void* d_ws, size_t ws_size,
                              hipStream_t stream) {
    const float* v_emb     = (const float*)d_in[0];
    const int*   batch_idx = (const int*)  d_in[1];
    const float* tokens_K  = (const float*)d_in[2];
    const float* tokens_V  = (const float*)d_in[3];
    const float* W_q       = (const float*)d_in[4];
    const float* b_q       = (const float*)d_in[5];
    float* out = (float*)d_out;

    int Nrows = in_sizes[1];   // batch_idx has N elements

    char*  ws     = (char*)d_ws;
    float* Wacc   = (float*)ws;                                 // 512*256 f32
    float* counts = (float*)(ws + (size_t)NSEG * NUM_TOKENS * 4);
    u16*   WqB    = (u16*)(ws + (size_t)NSEG * NUM_TOKENS * 4 + NSEG * 4);
    u16*   KtB    = WqB + 16384;

    hipMemsetAsync(Wacc, 0, (NSEG * NUM_TOKENS + NSEG) * sizeof(float), stream);

    pack_kernel<<<128, 256, 0, stream>>>(W_q, tokens_K, WqB, KtB);

    int nTiles = (Nrows + TILE_ROWS - 1) / TILE_ROWS;
    size_t smem = (size_t)(16384 * 3 + 32768) * 2           // bf16 buffers
                + (size_t)128 * NUM_TOKENS * 4              // scores
                + TOKEN_DIM * 4;                            // bias
    main_kernel<<<nTiles, 256, smem, stream>>>(v_emb, batch_idx, b_q,
                                               WqB, KtB, Wacc, counts, Nrows);

    finish_kernel<<<NSEG, TOKEN_DIM, 0, stream>>>(Wacc, counts, tokens_V, out);
}
